// Model_27393301413977
// MI455X (gfx1250) — compile-verified, matrix operand order checked by reading, not run on hardware
//
#include <hip/hip_runtime.h>
#include <hip/hip_bf16.h>

// ---------------------------------------------------------------------------
// Model dims (fixed by the reference)
// ---------------------------------------------------------------------------
#define DMODEL 512
#define NHEAD  8
#define NLAYER 6
#define DFF_   2048
#define VOCAB  32000
#define BATCH  2
#define SEQ    1024
#define DH_    64
#define ROWS   (BATCH * SEQ)   // 2048

typedef __attribute__((ext_vector_type(16))) __bf16 bf16x16;
typedef __attribute__((ext_vector_type(8)))  __bf16 bf16x8;
typedef __attribute__((ext_vector_type(8)))  float  f32x8;

#define TILE_M 128
#define TILE_N 64
#define TILE_K 32
#define LDS_STRIDE 40   // bf16 elems per LDS row: 80B, 16B-aligned, bank-friendly

__device__ __forceinline__ void cvt_store8(__bf16* dst, float4 a, float4 b)
{
    dst[0] = (__bf16)a.x; dst[1] = (__bf16)a.y;
    dst[2] = (__bf16)a.z; dst[3] = (__bf16)a.w;
    dst[4] = (__bf16)b.x; dst[5] = (__bf16)b.y;
    dst[6] = (__bf16)b.z; dst[7] = (__bf16)b.w;
}

__device__ __forceinline__ bf16x16 load_frag(const __bf16* base, int row, int hh)
{
    const bf16x8 lo = *(const bf16x8*)(base + row * LDS_STRIDE + hh * 8);
    const bf16x8 hi = *(const bf16x8*)(base + row * LDS_STRIDE + 16 + hh * 8);
    bf16x16 f;
    #pragma unroll
    for (int i = 0; i < 8; ++i) { f[i] = lo[i]; f[i + 8] = hi[i]; }
    return f;
}

// ---------------------------------------------------------------------------
// Generic batched GEMM:  C = act( alpha * A @ B + bias )
//   A: fp32 [M,K] row-major, lda ; B: fp32 [K,N] (btrans=0) or [N,K] (btrans=1)
//   Batched over grid.z: ptr += (z/H)*bs + (z%H)*hs for A,B,C.
// 128x64x32 block tile, 8 waves, each wave 32x32 (2 A-frag x 2 B-frag, 4 WMMA).
// ---------------------------------------------------------------------------
__global__ __launch_bounds__(256)
void gemm_wmma_bf16(const float* __restrict__ A, long long a_bs, long long a_hs, int lda,
                    const float* __restrict__ Bm, long long b_bs, long long b_hs, int ldb, int btrans,
                    float* __restrict__ C, long long c_bs, long long c_hs, int ldc,
                    const float* __restrict__ bias, float alpha, int relu,
                    int M, int N, int K, int nheads)
{
    __shared__ __bf16 As[TILE_M * LDS_STRIDE];   // [m][k]
    __shared__ __bf16 Bs[TILE_N * LDS_STRIDE];   // [n][k]

    const int z  = blockIdx.z;
    const int zb = z / nheads, zh = z % nheads;
    A  += (long long)zb * a_bs + (long long)zh * a_hs;
    Bm += (long long)zb * b_bs + (long long)zh * b_hs;
    C  += (long long)zb * c_bs + (long long)zh * c_hs;

    const int m0 = blockIdx.y * TILE_M;
    const int n0 = blockIdx.x * TILE_N;

    const int tid  = threadIdx.x;
    const int wave = tid >> 5;
    const int lane = tid & 31;
    const int lm   = lane & 15;   // row (A) / col (B,C) within 16
    const int hh   = lane >> 4;   // half-wave selects K sub-range / M group
    const int mw   = (wave & 3) * 32;
    const int nw   = (wave >> 2) * 32;

    f32x8 acc00 = {}, acc01 = {}, acc10 = {}, acc11 = {};

    for (int k0 = 0; k0 < K; k0 += TILE_K) {
        // ---- stage A tile 128x32 fp32 -> bf16 LDS (16 elems/thread) ----
        {
            const int r = tid >> 1;           // 0..127
            const int c = (tid & 1) * 16;     // 0 or 16
            const float4* src = (const float4*)(A + (long long)(m0 + r) * lda + k0 + c);
            float4 v0 = src[0], v1 = src[1], v2 = src[2], v3 = src[3];
            if (k0 + TILE_K < K)
                __builtin_prefetch(A + (long long)(m0 + r) * lda + (k0 + TILE_K) + c, 0, 1);
            __bf16* dst = As + r * LDS_STRIDE + c;
            cvt_store8(dst,     v0, v1);
            cvt_store8(dst + 8, v2, v3);
        }
        // ---- stage B tile -> Bs[n][k] (8 elems/thread) ----
        if (btrans) { // global B is [N,K]: direct copy
            const int r = tid >> 2;          // n: 0..63
            const int c = (tid & 3) * 8;     // k: 0..31 step 8
            const float4* src = (const float4*)(Bm + (long long)(n0 + r) * ldb + k0 + c);
            float4 v0 = src[0], v1 = src[1];
            if (k0 + TILE_K < K)
                __builtin_prefetch(Bm + (long long)(n0 + r) * ldb + (k0 + TILE_K) + c, 0, 1);
            cvt_store8(Bs + r * LDS_STRIDE + c, v0, v1);
        } else {      // global B is [K,N]: transpose into LDS
            const int r = tid >> 3;          // k: 0..31
            const int c = (tid & 7) * 8;     // n: 0..56 step 8
            const float4* src = (const float4*)(Bm + (long long)(k0 + r) * ldb + n0 + c);
            float4 v0 = src[0], v1 = src[1];
            if (k0 + TILE_K < K)
                __builtin_prefetch(Bm + (long long)(k0 + TILE_K + r) * ldb + n0 + c, 0, 1);
            float vv[8] = {v0.x, v0.y, v0.z, v0.w, v1.x, v1.y, v1.z, v1.w};
            #pragma unroll
            for (int i = 0; i < 8; ++i)
                Bs[(c + i) * LDS_STRIDE + r] = (__bf16)vv[i];
        }
        __syncthreads();

        // ---- fragments: 2xA, 2xB -> 4 WMMA (per-lane layout per ISA tables) ----
        const bf16x16 a0 = load_frag(As, mw + lm,      hh);
        const bf16x16 a1 = load_frag(As, mw + 16 + lm, hh);
        const bf16x16 b0 = load_frag(Bs, nw + lm,      hh);
        const bf16x16 b1 = load_frag(Bs, nw + 16 + lm, hh);

        acc00 = __builtin_amdgcn_wmma_f32_16x16x32_bf16(false, a0, false, b0,
                                                        (short)0, acc00, false, false);
        acc01 = __builtin_amdgcn_wmma_f32_16x16x32_bf16(false, a0, false, b1,
                                                        (short)0, acc01, false, false);
        acc10 = __builtin_amdgcn_wmma_f32_16x16x32_bf16(false, a1, false, b0,
                                                        (short)0, acc10, false, false);
        acc11 = __builtin_amdgcn_wmma_f32_16x16x32_bf16(false, a1, false, b1,
                                                        (short)0, acc11, false, false);
        __syncthreads();
    }

    // ---- epilogue: C layout VGPR r -> m = r + 8*h, lane&15 -> n ----
    #pragma unroll
    for (int r = 0; r < 8; ++r) {
        const int ma = m0 + mw + hh * 8 + r;        // A-frag 0 rows
        const int mb = ma + 16;                     // A-frag 1 rows
        const int c0 = n0 + nw + lm;                // B-frag 0 cols
        const int c1 = c0 + 16;                     // B-frag 1 cols
        float v;
        v = acc00[r] * alpha; if (bias) v += bias[c0]; if (relu) v = fmaxf(v, 0.0f);
        C[(long long)ma * ldc + c0] = v;
        v = acc01[r] * alpha; if (bias) v += bias[c1]; if (relu) v = fmaxf(v, 0.0f);
        C[(long long)ma * ldc + c1] = v;
        v = acc10[r] * alpha; if (bias) v += bias[c0]; if (relu) v = fmaxf(v, 0.0f);
        C[(long long)mb * ldc + c0] = v;
        v = acc11[r] * alpha; if (bias) v += bias[c1]; if (relu) v = fmaxf(v, 0.0f);
        C[(long long)mb * ldc + c1] = v;
    }
}

// ---------------------------------------------------------------------------
// Embedding * sqrt(D) + sinusoidal positional encoding
// ---------------------------------------------------------------------------
__global__ __launch_bounds__(256)
void embed_pe(const int* __restrict__ tok, const float* __restrict__ emb,
              float* __restrict__ out)
{
    const int row = blockIdx.x;
    const int s   = row % SEQ;
    const int t   = tok[row];
    const float scale = 22.62741699796952f;          // sqrt(512)
    for (int d = threadIdx.x; d < DMODEL; d += 256) {
        const float i2  = (float)(d & ~1);
        const float ang = (float)s * __expf(-i2 * (9.210340371976184f / (float)DMODEL));
        const float pe  = (d & 1) ? __cosf(ang) : __sinf(ang);
        out[(long long)row * DMODEL + d] = emb[(long long)t * DMODEL + d] * scale + pe;
    }
}

// ---------------------------------------------------------------------------
// out = LayerNorm(x + r) * gamma + beta      (row length DMODEL=512)
// ---------------------------------------------------------------------------
__global__ __launch_bounds__(128)
void add_layernorm(const float* __restrict__ x, const float* __restrict__ r,
                   const float* __restrict__ gamma, const float* __restrict__ beta,
                   float* __restrict__ out)
{
    const long long row = blockIdx.x;
    const int tid = threadIdx.x;
    const float* xr = x + row * DMODEL;
    const float* rr = r + row * DMODEL;

    float v[4];
    float s = 0.0f;
    #pragma unroll
    for (int i = 0; i < 4; ++i) {
        const int c = tid + i * 128;
        v[i] = xr[c] + rr[c];
        s += v[i];
    }
    __shared__ float red[128];
    red[tid] = s; __syncthreads();
    for (int k = 64; k > 0; k >>= 1) { if (tid < k) red[tid] += red[tid + k]; __syncthreads(); }
    const float mean = red[0] * (1.0f / DMODEL);
    __syncthreads();

    float s2 = 0.0f;
    #pragma unroll
    for (int i = 0; i < 4; ++i) { const float d = v[i] - mean; s2 += d * d; }
    red[tid] = s2; __syncthreads();
    for (int k = 64; k > 0; k >>= 1) { if (tid < k) red[tid] += red[tid + k]; __syncthreads(); }
    const float inv = rsqrtf(red[0] * (1.0f / DMODEL) + 1e-5f);
    __syncthreads();

    #pragma unroll
    for (int i = 0; i < 4; ++i) {
        const int c = tid + i * 128;
        out[row * DMODEL + c] = (v[i] - mean) * inv * gamma[c] + beta[c];
    }
}

// ---------------------------------------------------------------------------
// Masked softmax over Sk=1024 columns, in place.
//   causal=1 : mask col > row ; causal=0 : mask where src_tok[b, col]==PAD(0)
// ---------------------------------------------------------------------------
__global__ __launch_bounds__(256)
void softmax_mask(float* __restrict__ scores, const int* __restrict__ src_tok,
                  int causal, int Sq, int Sk, int nheads)
{
    const int q = blockIdx.x;
    const int z = blockIdx.y;
    const int b = z / nheads;
    float* row = scores + ((long long)z * Sq + q) * (long long)Sk;
    const int tid = threadIdx.x;

    float v[4];
    float mx = -3.4e38f;
    #pragma unroll
    for (int i = 0; i < 4; ++i) {
        const int c = tid + i * 256;
        float x = row[c];
        const bool masked = causal ? (c > q) : (src_tok[(long long)b * Sk + c] == 0);
        x = masked ? -1e9f : x;
        v[i] = x;
        mx = fmaxf(mx, x);
    }
    __shared__ float red[256];
    red[tid] = mx; __syncthreads();
    for (int k = 128; k > 0; k >>= 1) { if (tid < k) red[tid] = fmaxf(red[tid], red[tid + k]); __syncthreads(); }
    mx = red[0]; __syncthreads();

    float sum = 0.0f;
    #pragma unroll
    for (int i = 0; i < 4; ++i) { v[i] = __expf(v[i] - mx); sum += v[i]; }
    red[tid] = sum; __syncthreads();
    for (int k = 128; k > 0; k >>= 1) { if (tid < k) red[tid] += red[tid + k]; __syncthreads(); }
    const float inv = 1.0f / red[0];
    #pragma unroll
    for (int i = 0; i < 4; ++i) row[tid + i * 256] = v[i] * inv;
}

// ---------------------------------------------------------------------------
// Host orchestration
// ---------------------------------------------------------------------------
extern "C" void kernel_launch(void* const* d_in, const int* in_sizes, int n_in,
                              void* d_out, int out_size, void* d_ws, size_t ws_size,
                              hipStream_t stream)
{
    const int*   batch_src = (const int*)d_in[0];
    const int*   trg_teach = (const int*)d_in[1];
    const float* src_emb   = (const float*)d_in[2];
    const float* trg_emb   = (const float*)d_in[3];
    const float* fc_w      = (const float*)d_in[4];
    const float* fc_b      = (const float*)d_in[5];
    const float* enc_wqkv  = (const float*)d_in[6];
    const float* enc_wo    = (const float*)d_in[7];
    const float* enc_ln1s  = (const float*)d_in[8];
    const float* enc_ln1b  = (const float*)d_in[9];
    const float* enc_w1    = (const float*)d_in[10];
    const float* enc_b1    = (const float*)d_in[11];
    const float* enc_w2    = (const float*)d_in[12];
    const float* enc_b2    = (const float*)d_in[13];
    const float* enc_ln2s  = (const float*)d_in[14];
    const float* enc_ln2b  = (const float*)d_in[15];
    const float* dec_wqkv  = (const float*)d_in[16];
    const float* dec_wo    = (const float*)d_in[17];
    const float* dec_ln1s  = (const float*)d_in[18];
    const float* dec_ln1b  = (const float*)d_in[19];
    const float* dec_wq    = (const float*)d_in[20];
    const float* dec_wkv   = (const float*)d_in[21];
    const float* dec_woc   = (const float*)d_in[22];
    const float* dec_ln2s  = (const float*)d_in[23];
    const float* dec_ln2b  = (const float*)d_in[24];
    const float* dec_w1    = (const float*)d_in[25];
    const float* dec_b1    = (const float*)d_in[26];
    const float* dec_w2    = (const float*)d_in[27];
    const float* dec_b2    = (const float*)d_in[28];
    const float* dec_ln3s  = (const float*)d_in[29];
    const float* dec_ln3b  = (const float*)d_in[30];

    // workspace layout (fp32)
    float* ws  = (float*)d_ws;
    float* X   = ws;                                         // [2048,512] enc activations / enc_out
    float* Y   = X   + (size_t)ROWS * DMODEL;                // [2048,512] dec activations
    float* QKV = Y   + (size_t)ROWS * DMODEL;                // [2048,1536] qkv / cross kv
    float* SC  = QKV + (size_t)ROWS * 3 * DMODEL;            // [16,1024,1024] scores
    float* AO  = SC  + (size_t)BATCH * NHEAD * SEQ * SEQ;    // [2048,512] merged attn out
    float* T1  = AO  + (size_t)ROWS * DMODEL;                // [2048,512] proj tmp
    float* FF  = T1  + (size_t)ROWS * DMODEL;                // [2048,2048] ffn hidden
    float* QC  = FF  + (size_t)ROWS * DFF_;                  // [2048,512] cross queries

    const long long QKV_BS = (long long)SEQ * 3 * DMODEL;    // per-batch stride in QKV
    const long long SC_BS  = (long long)NHEAD * SEQ * SEQ;
    const long long SC_HS  = (long long)SEQ * SEQ;

    auto gemm = [&](const float* A, long long abs_, long long ahs, int lda,
                    const float* Bm, long long bbs, long long bhs, int ldb, int btrans,
                    float* C, long long cbs, long long chs, int ldc,
                    const float* bias, float alpha, int relu,
                    int M, int N, int K, int Z) {
        dim3 g(N / TILE_N, M / TILE_M, Z);
        gemm_wmma_bf16<<<g, 256, 0, stream>>>(A, abs_, ahs, lda, Bm, bbs, bhs, ldb, btrans,
                                              C, cbs, chs, ldc, bias, alpha, relu,
                                              M, N, K, NHEAD);
    };

    // ===================== encoder =====================
    embed_pe<<<ROWS, 256, 0, stream>>>(batch_src, src_emb, X);

    for (int l = 0; l < NLAYER; ++l) {
        const float* wqkv = enc_wqkv + (size_t)l * DMODEL * 3 * DMODEL;
        // qkv projection
        gemm(X, 0, 0, DMODEL, wqkv, 0, 0, 3 * DMODEL, 0,
             QKV, 0, 0, 3 * DMODEL, nullptr, 1.0f, 0, ROWS, 3 * DMODEL, DMODEL, 1);
        // scores = Q @ K^T * 1/sqrt(dh)   (batched over B*H)
        gemm(QKV,          QKV_BS, 64, 3 * DMODEL,
             QKV + DMODEL, QKV_BS, 64, 3 * DMODEL, 1,
             SC, SC_BS, SC_HS, SEQ, nullptr, 0.125f, 0, SEQ, SEQ, DH_, BATCH * NHEAD);
        softmax_mask<<<dim3(SEQ, BATCH * NHEAD), 256, 0, stream>>>(
            SC, batch_src, 0, SEQ, SEQ, NHEAD);
        // O = P @ V  (merged-head output)
        gemm(SC, SC_BS, SC_HS, SEQ,
             QKV + 2 * DMODEL, QKV_BS, 64, 3 * DMODEL, 0,
             AO, (long long)SEQ * DMODEL, 64, DMODEL,
             nullptr, 1.0f, 0, SEQ, DH_, SEQ, BATCH * NHEAD);
        // output projection
        gemm(AO, 0, 0, DMODEL, enc_wo + (size_t)l * DMODEL * DMODEL, 0, 0, DMODEL, 0,
             T1, 0, 0, DMODEL, nullptr, 1.0f, 0, ROWS, DMODEL, DMODEL, 1);
        add_layernorm<<<ROWS, 128, 0, stream>>>(X, T1, enc_ln1s + l * DMODEL,
                                                enc_ln1b + l * DMODEL, X);
        // FFN
        gemm(X, 0, 0, DMODEL, enc_w1 + (size_t)l * DMODEL * DFF_, 0, 0, DFF_, 0,
             FF, 0, 0, DFF_, enc_b1 + (size_t)l * DFF_, 1.0f, 1, ROWS, DFF_, DMODEL, 1);
        gemm(FF, 0, 0, DFF_, enc_w2 + (size_t)l * DFF_ * DMODEL, 0, 0, DMODEL, 0,
             T1, 0, 0, DMODEL, enc_b2 + (size_t)l * DMODEL, 1.0f, 0, ROWS, DMODEL, DFF_, 1);
        add_layernorm<<<ROWS, 128, 0, stream>>>(X, T1, enc_ln2s + l * DMODEL,
                                                enc_ln2b + l * DMODEL, X);
    }

    // ===================== decoder =====================
    embed_pe<<<ROWS, 256, 0, stream>>>(trg_teach, trg_emb, Y);

    for (int l = 0; l < NLAYER; ++l) {
        const float* wqkv = dec_wqkv + (size_t)l * DMODEL * 3 * DMODEL;
        // ---- masked self-attention ----
        gemm(Y, 0, 0, DMODEL, wqkv, 0, 0, 3 * DMODEL, 0,
             QKV, 0, 0, 3 * DMODEL, nullptr, 1.0f, 0, ROWS, 3 * DMODEL, DMODEL, 1);
        gemm(QKV,          QKV_BS, 64, 3 * DMODEL,
             QKV + DMODEL, QKV_BS, 64, 3 * DMODEL, 1,
             SC, SC_BS, SC_HS, SEQ, nullptr, 0.125f, 0, SEQ, SEQ, DH_, BATCH * NHEAD);
        softmax_mask<<<dim3(SEQ, BATCH * NHEAD), 256, 0, stream>>>(
            SC, batch_src, 1, SEQ, SEQ, NHEAD);
        gemm(SC, SC_BS, SC_HS, SEQ,
             QKV + 2 * DMODEL, QKV_BS, 64, 3 * DMODEL, 0,
             AO, (long long)SEQ * DMODEL, 64, DMODEL,
             nullptr, 1.0f, 0, SEQ, DH_, SEQ, BATCH * NHEAD);
        gemm(AO, 0, 0, DMODEL, dec_wo + (size_t)l * DMODEL * DMODEL, 0, 0, DMODEL, 0,
             T1, 0, 0, DMODEL, nullptr, 1.0f, 0, ROWS, DMODEL, DMODEL, 1);
        add_layernorm<<<ROWS, 128, 0, stream>>>(Y, T1, dec_ln1s + l * DMODEL,
                                                dec_ln1b + l * DMODEL, Y);
        // ---- cross-attention ----
        gemm(Y, 0, 0, DMODEL, dec_wq + (size_t)l * DMODEL * DMODEL, 0, 0, DMODEL, 0,
             QC, 0, 0, DMODEL, nullptr, 1.0f, 0, ROWS, DMODEL, DMODEL, 1);
        // kc,vc = enc_out @ wkv  -> reuse QKV buffer ([2048,1024])
        gemm(X, 0, 0, DMODEL, dec_wkv + (size_t)l * DMODEL * 2 * DMODEL, 0, 0, 2 * DMODEL, 0,
             QKV, 0, 0, 2 * DMODEL, nullptr, 1.0f, 0, ROWS, 2 * DMODEL, DMODEL, 1);
        gemm(QC,  (long long)SEQ * DMODEL,     64, DMODEL,
             QKV, (long long)SEQ * 2 * DMODEL, 64, 2 * DMODEL, 1,
             SC, SC_BS, SC_HS, SEQ, nullptr, 0.125f, 0, SEQ, SEQ, DH_, BATCH * NHEAD);
        softmax_mask<<<dim3(SEQ, BATCH * NHEAD), 256, 0, stream>>>(
            SC, batch_src, 0, SEQ, SEQ, NHEAD);
        gemm(SC, SC_BS, SC_HS, SEQ,
             QKV + DMODEL, (long long)SEQ * 2 * DMODEL, 64, 2 * DMODEL, 0,
             AO, (long long)SEQ * DMODEL, 64, DMODEL,
             nullptr, 1.0f, 0, SEQ, DH_, SEQ, BATCH * NHEAD);
        gemm(AO, 0, 0, DMODEL, dec_woc + (size_t)l * DMODEL * DMODEL, 0, 0, DMODEL, 0,
             T1, 0, 0, DMODEL, nullptr, 1.0f, 0, ROWS, DMODEL, DMODEL, 1);
        add_layernorm<<<ROWS, 128, 0, stream>>>(Y, T1, dec_ln2s + l * DMODEL,
                                                dec_ln2b + l * DMODEL, Y);
        // ---- FFN ----
        gemm(Y, 0, 0, DMODEL, dec_w1 + (size_t)l * DMODEL * DFF_, 0, 0, DFF_, 0,
             FF, 0, 0, DFF_, dec_b1 + (size_t)l * DFF_, 1.0f, 1, ROWS, DFF_, DMODEL, 1);
        gemm(FF, 0, 0, DFF_, dec_w2 + (size_t)l * DFF_ * DMODEL, 0, 0, DMODEL, 0,
             T1, 0, 0, DMODEL, dec_b2 + (size_t)l * DMODEL, 1.0f, 0, ROWS, DMODEL, DFF_, 1);
        add_layernorm<<<ROWS, 128, 0, stream>>>(Y, T1, dec_ln3s + l * DMODEL,
                                                dec_ln3b + l * DMODEL, Y);
    }

    // ===================== logits =====================
    gemm(Y, 0, 0, DMODEL, fc_w, 0, 0, VOCAB, 0,
         (float*)d_out, 0, 0, VOCAB, fc_b, 1.0f, 0, ROWS, VOCAB, DMODEL, 1);
}
